// MambaBlock_45105746542916
// MI455X (gfx1250) — matrix operand order repaired and checked by reading, not compiled, not run on hardware
//
#include <hip/hip_runtime.h>
#include <hip/hip_bf16.h>
#include <math.h>

typedef __bf16 bf16;
typedef __bf16 v16bf __attribute__((ext_vector_type(16)));
typedef float  v8f   __attribute__((ext_vector_type(8)));
typedef int    v4i_t __attribute__((vector_size(16)));

#define DMODEL  512
#define DINNER  1024
#define DSTATE  128
#define CONVDIM 1280
#define DPROJ   2305
#define BATCH   8
#define SEQ     2048
#define MTOT    (BATCH*SEQ)    // 16384
#define EPSR    1e-5f

#define AS1 __attribute__((address_space(1)))
#define AS3 __attribute__((address_space(3)))

#if defined(__HIP_DEVICE_COMPILE__) && __has_builtin(__builtin_amdgcn_global_load_async_to_lds_b128)
#define HAVE_ASYNC 1
#else
#define HAVE_ASYNC 0
#endif

// 16-byte global -> LDS copy: async (no VGPR round trip, ASYNCcnt) if available
__device__ __forceinline__ void cp16_g2l(bf16* ldst, const bf16* gsrc) {
#if HAVE_ASYNC
    __builtin_amdgcn_global_load_async_to_lds_b128(
        (AS1 v4i_t*)gsrc, (AS3 v4i_t*)ldst, 0, 0);
#else
    *(uint4*)ldst = *(const uint4*)gsrc;
#endif
}
__device__ __forceinline__ void cp_wait() {
#if HAVE_ASYNC
#if __has_builtin(__builtin_amdgcn_s_wait_asynccnt)
    __builtin_amdgcn_s_wait_asynccnt(0);
#else
    asm volatile("s_wait_asynccnt 0x0" ::: "memory");
#endif
#endif
}

// ---------------------------------------------------------------- reductions
__device__ __forceinline__ float blockReduceSum256(float v) {
    __shared__ float s[8];
    int lane = threadIdx.x & 31, w = threadIdx.x >> 5;
#pragma unroll
    for (int o = 16; o > 0; o >>= 1) v += __shfl_xor(v, o, 32);
    if (lane == 0) s[w] = v;
    __syncthreads();
    float t = 0.f;
#pragma unroll
    for (int i = 0; i < 8; i++) t += s[i];
    __syncthreads();
    return t;
}

// ---------------------------------------------------------------- elementwise
__global__ __launch_bounds__(256) void f32_to_bf16_kernel(const float* __restrict__ in,
                                                          bf16* __restrict__ out, int n) {
    int i = blockIdx.x * 256 + threadIdx.x;
    if (i < n) out[i] = (bf16)in[i];
}

__global__ __launch_bounds__(256) void copy_f32_kernel(const float* __restrict__ in,
                                                       float* __restrict__ out, int n) {
    int i = blockIdx.x * 256 + threadIdx.x;
    if (i < n) out[i] = in[i];
}

// rmsnorm over width 512, f32 in -> bf16 out (one block per row)
__global__ __launch_bounds__(256) void rmsnorm512_bf16_kernel(const float* __restrict__ x,
                                                              const float* __restrict__ w,
                                                              bf16* __restrict__ out) {
    int row = blockIdx.x;
    const float* xr = x + (size_t)row * DMODEL;
    float v0 = xr[threadIdx.x], v1 = xr[threadIdx.x + 256];
    float ss = blockReduceSum256(v0 * v0 + v1 * v1);
    float rs = rsqrtf(ss * (1.0f / DMODEL) + EPSR);
    out[(size_t)row * DMODEL + threadIdx.x]       = (bf16)(v0 * rs * w[threadIdx.x]);
    out[(size_t)row * DMODEL + threadIdx.x + 256] = (bf16)(v1 * rs * w[threadIdx.x + 256]);
}

// dt = softplus(zxbcdt[:, 2304] + dt_bias[0])
__global__ __launch_bounds__(256) void dt_kernel(const float* __restrict__ zxbcdt,
                                                 const float* __restrict__ dt_bias,
                                                 float* __restrict__ dt) {
    int i = blockIdx.x * 256 + threadIdx.x;
    if (i < MTOT) {
        float v = zxbcdt[(size_t)i * DPROJ + (DPROJ - 1)] + dt_bias[0];
        dt[i] = (v > 20.f) ? v : log1pf(expf(v));
    }
}

// depthwise causal conv (width 4) + silu over the xBC channels of zxbcdt
__global__ __launch_bounds__(256) void conv_silu_kernel(const float* __restrict__ zxbcdt,
                                                        const float* __restrict__ cw,
                                                        const float* __restrict__ cb,
                                                        float* __restrict__ out) {
    int row = blockIdx.x;            // b*SEQ + l
    int l = row & (SEQ - 1);
    for (int c = threadIdx.x; c < CONVDIM; c += 256) {
        float acc = cb[c];
#pragma unroll
        for (int j = 0; j < 4; j++) {
            int ll = l - 3 + j;
            if (ll >= 0)
                acc += cw[c * 4 + j] * zxbcdt[(size_t)(row - 3 + j) * DPROJ + DINNER + c];
        }
        out[(size_t)row * CONVDIM + c] = acc / (1.f + expf(-acc));   // silu
    }
}

// ---------------------------------------------------------------- scan
// grid = (BATCH, DINNER/64); 256 threads; thread (p_local = t>>2, q = t&3) owns
// 32 state floats h[p, q*32 .. q*32+31] in VGPRs for the whole sequence.
// B/C are double-buffered in LDS -> one barrier per timestep.
__global__ __launch_bounds__(256) void scan_kernel(const float* __restrict__ conv_out,
                                                   const float* __restrict__ dtv,
                                                   const float* __restrict__ A_log,
                                                   const float* __restrict__ Dp,
                                                   float* __restrict__ ys) {
    __shared__ float sB[2][DSTATE];
    __shared__ float sC[2][DSTATE];
    int b  = blockIdx.x;
    int pl = threadIdx.x >> 2;
    int q  = threadIdx.x & 3;
    int p  = blockIdx.y * 64 + pl;
    int n0 = q * 32;
    float A  = -expf(A_log[0]);
    float D0 = Dp[0];
    float h[32];
#pragma unroll
    for (int j = 0; j < 32; j++) h[j] = 0.f;
    size_t rowbase = (size_t)b * SEQ;
    for (int l = 0; l < SEQ; l++) {
        int pb = l & 1;
        size_t row = rowbase + l;
        const float* cr = conv_out + row * CONVDIM;
        if (threadIdx.x < 128) sB[pb][threadIdx.x]       = cr[DINNER + threadIdx.x];
        else                   sC[pb][threadIdx.x - 128] = cr[DINNER + threadIdx.x];
        float dt = dtv[row];
        float xp = cr[p];
        if (l + 1 < SEQ)
            __builtin_prefetch((const void*)(cr + CONVDIM + DINNER + threadIdx.x), 0, 1);
        __syncthreads();
        float dA  = expf(dt * A);
        float dtx = dt * xp;
        float yp = 0.f;
#pragma unroll
        for (int j = 0; j < 32; j++) {
            h[j] = h[j] * dA + dtx * sB[pb][n0 + j];
            yp  += h[j] * sC[pb][n0 + j];
        }
        yp += __shfl_xor(yp, 1, 32);
        yp += __shfl_xor(yp, 2, 32);
        if (q == 0) ys[row * DINNER + p] = yp + D0 * xp;
        // no trailing barrier: next step writes the other buffer; reuse of this
        // buffer is fenced by the barrier of step l+1.
    }
}

// gated rmsnorm: yg = ys * silu(z); rmsnorm(yg)*gw -> bf16 (one block per row)
__global__ __launch_bounds__(256) void gate_norm_kernel(const float* __restrict__ ys,
                                                        const float* __restrict__ zxbcdt,
                                                        const float* __restrict__ gw,
                                                        bf16* __restrict__ out) {
    int row = blockIdx.x;
    const float* yr = ys + (size_t)row * DINNER;
    const float* zr = zxbcdt + (size_t)row * DPROJ;   // z = cols [0,1024)
    float v[4];
    float ss = 0.f;
#pragma unroll
    for (int i = 0; i < 4; i++) {
        int c = threadIdx.x + i * 256;
        float z = zr[c];
        float g = yr[c] * (z / (1.f + expf(-z)));
        v[i] = g;
        ss += g * g;
    }
    ss = blockReduceSum256(ss);
    float rs = rsqrtf(ss * (1.0f / DINNER) + EPSR);
#pragma unroll
    for (int i = 0; i < 4; i++) {
        int c = threadIdx.x + i * 256;
        out[(size_t)row * DINNER + c] = (bf16)(v[i] * rs * gw[c]);
    }
}

// ---------------------------------------------------------------- WMMA GEMM
// out(M,N) = A(M,K) * W(N,K)^T ; A,W bf16, acc f32.
// Block tile 128x128, 8 waves (wave w -> rows w*16..w*16+15, all 8 N-subtiles).
// W slab (128 rows x 32 K halves) double-buffered in LDS (40-half pitch,
// conflict-free b128 reads); slab k+1 streamed via async global->LDS while
// computing slab k; all 16 B-fragment ds_loads issued before the 8 WMMAs.
// EPI: 0 = f32 store, 1 = f32 store + addsrc, 2 = exact GELU -> bf16 store.
#define LDSP 40
template <int EPI>
__global__ __launch_bounds__(256) void gemm_wmma_kernel(const bf16* __restrict__ A,
                                                        const bf16* __restrict__ W,
                                                        int M, int N, int K,
                                                        float* __restrict__ outF,
                                                        bf16* __restrict__ outB,
                                                        const float* __restrict__ addsrc) {
    __shared__ bf16 sB[2][128 * LDSP];
    int nblk = blockIdx.x * 128;
    int mblk = blockIdx.y * 128;
    int wave = threadIdx.x >> 5;
    int lane = threadIdx.x & 31;
    int hi   = lane >> 4;            // 0/1
    int col0 = lane & 15;
    int mrow = mblk + wave * 16 + col0;
    int kqA  = hi * 8;               // A-fragment K sub-offset
    int kqB  = hi * 8;               // B-fragment K sub-offset

    v8f acc[8];
#pragma unroll
    for (int i = 0; i < 8; i++)
#pragma unroll
        for (int j = 0; j < 8; j++) acc[i][j] = 0.f;

    // cooperative slab loader: 128 rows x 32 halves (4x16B per row)
    auto load_slab = [&](bf16* dstbuf, int k0) {
        for (int idx = threadIdx.x; idx < 128 * 4; idx += 256) {
            int rn = idx >> 2, qq = idx & 3;
            int ng = nblk + rn;
            bf16* dst = &dstbuf[rn * LDSP + qq * 8];
            if (ng < N) cp16_g2l(dst, W + (size_t)ng * K + k0 + qq * 8);
            else        *(uint4*)dst = make_uint4(0u, 0u, 0u, 0u);
        }
    };

    load_slab(sB[0], 0);
    cp_wait();
    __syncthreads();

    for (int k0 = 0; k0 < K; k0 += 32) {
        int cb = (k0 >> 5) & 1;
        bf16* cur = sB[cb];
        if (k0 + 32 < K) load_slab(sB[cb ^ 1], k0 + 32);   // pipeline next slab

        // A fragment: lane(hi,m): K {kqA..kqA+7} U {kqA+16..kqA+23}
        union { v16bf v; uint4 u[2]; } afr;
        const bf16* ap = A + (size_t)mrow * K + k0 + kqA;
        afr.u[0] = *(const uint4*)(ap);
        afr.u[1] = *(const uint4*)(ap + 16);
        if (k0 + 32 < K) __builtin_prefetch((const void*)(ap + 32), 0, 1);

        // issue all 16 B-fragment ds_load_b128 first ...
        union { v16bf v; uint4 u[2]; } bfr[8];
#pragma unroll
        for (int ns = 0; ns < 8; ns++) {
            const bf16* bp = &cur[(ns * 16 + col0) * LDSP + kqB];
            bfr[ns].u[0] = *(const uint4*)(bp);
            bfr[ns].u[1] = *(const uint4*)(bp + 16);
        }
        // ... then 8 back-to-back WMMAs
#pragma unroll
        for (int ns = 0; ns < 8; ns++)
            acc[ns] = __builtin_amdgcn_wmma_f32_16x16x32_bf16(
                false, afr.v, false, bfr[ns].v, (short)0, acc[ns], false, false);

        cp_wait();        // my async fills of the next slab are done
        __syncthreads();  // everyone done reading cur / filling next
    }

    // epilogue: VGPR r <-> row r + 8*hi, col = lane&15
#pragma unroll
    for (int ns = 0; ns < 8; ns++) {
        int cg = nblk + ns * 16 + col0;
        if (cg >= N) continue;
#pragma unroll
        for (int r = 0; r < 8; r++) {
            int rg = mblk + wave * 16 + r + 8 * hi;
            float v = acc[ns][r];
            size_t o = (size_t)rg * N + cg;
            if (EPI == 0) {
                outF[o] = v;
            } else if (EPI == 1) {
                outF[o] = v + addsrc[o];
            } else {
                float g = 0.5f * v * (1.0f + erff(v * 0.70710678118f));
                outB[o] = (bf16)g;
            }
        }
    }
}

// ---------------------------------------------------------------- launch
extern "C" void kernel_launch(void* const* d_in, const int* in_sizes, int n_in,
                              void* d_out, int out_size, void* d_ws, size_t ws_size,
                              hipStream_t stream) {
    const float* x          = (const float*)d_in[0];
    const float* rnn_state  = (const float*)d_in[2];
    const float* norm1_w    = (const float*)d_in[3];
    const float* in_proj_w  = (const float*)d_in[4];
    const float* conv_w     = (const float*)d_in[5];
    const float* conv_b     = (const float*)d_in[6];
    const float* dt_bias    = (const float*)d_in[7];
    const float* A_log      = (const float*)d_in[8];
    const float* Dp         = (const float*)d_in[9];
    const float* gated_w    = (const float*)d_in[10];
    const float* out_proj_w = (const float*)d_in[11];
    const float* norm2_w    = (const float*)d_in[12];
    const float* mlp_w1     = (const float*)d_in[13];
    const float* mlp_w2     = (const float*)d_in[14];
    float* out = (float*)d_out;

    // workspace carve-up (256B aligned)
    char* ws = (char*)d_ws;
    size_t off = 0;
    auto carve = [&](size_t bytes) -> char* {
        char* p = ws + off;
        off = (off + bytes + 255) & ~(size_t)255;
        return p;
    };
    bf16*  u_bf     = (bf16*)carve((size_t)MTOT * DMODEL * 2);
    bf16*  w_in_bf  = (bf16*)carve((size_t)DPROJ * DMODEL * 2);
    bf16*  w_out_bf = (bf16*)carve((size_t)DMODEL * DINNER * 2);
    bf16*  w_m1_bf  = (bf16*)carve((size_t)(4 * DMODEL) * DMODEL * 2);
    bf16*  w_m2_bf  = (bf16*)carve((size_t)DMODEL * (4 * DMODEL) * 2);
    float* zxbcdt   = (float*)carve((size_t)MTOT * DPROJ * 4);
    float* conv_out = (float*)carve((size_t)MTOT * CONVDIM * 4);
    float* dtbuf    = (float*)carve((size_t)MTOT * 4);
    float* ysbuf    = (float*)carve((size_t)MTOT * DINNER * 4);
    bf16*  ygate_bf = (bf16*)carve((size_t)MTOT * DINNER * 2);
    // aliases (lifetimes are disjoint):
    float* resid    = conv_out;          // 33.5MB <= 84MB, conv_out dead after scan
    bf16*  h2_bf    = (bf16*)ysbuf;      // 16.7MB <= 67MB, ys dead after gating
    bf16*  gelu_bf  = (bf16*)zxbcdt;     // 67MB <= 151MB, z dead after gating
    (void)ws_size; (void)n_in; (void)in_sizes; (void)out_size;

    // weights -> bf16
    {
        int n1 = DPROJ * DMODEL, n2 = DMODEL * DINNER, n3 = 4 * DMODEL * DMODEL,
            n4 = DMODEL * 4 * DMODEL;
        f32_to_bf16_kernel<<<(n1 + 255) / 256, 256, 0, stream>>>(in_proj_w, w_in_bf, n1);
        f32_to_bf16_kernel<<<(n2 + 255) / 256, 256, 0, stream>>>(out_proj_w, w_out_bf, n2);
        f32_to_bf16_kernel<<<(n3 + 255) / 256, 256, 0, stream>>>(mlp_w1, w_m1_bf, n3);
        f32_to_bf16_kernel<<<(n4 + 255) / 256, 256, 0, stream>>>(mlp_w2, w_m2_bf, n4);
    }
    // 1) u = rmsnorm(x, norm1_w) -> bf16
    rmsnorm512_bf16_kernel<<<MTOT, 256, 0, stream>>>(x, norm1_w, u_bf);
    // 2) zxbcdt = u @ in_proj_w^T   (16384 x 2305, K=512)
    gemm_wmma_kernel<0><<<dim3((DPROJ + 127) / 128, MTOT / 128), 256, 0, stream>>>(
        u_bf, w_in_bf, MTOT, DPROJ, DMODEL, zxbcdt, nullptr, nullptr);
    // 3) dt = softplus(dt + bias)
    dt_kernel<<<(MTOT + 255) / 256, 256, 0, stream>>>(zxbcdt, dt_bias, dtbuf);
    // 4) xBC = silu(causal depthwise conv(xBC))
    conv_silu_kernel<<<MTOT, 256, 0, stream>>>(zxbcdt, conv_w, conv_b, conv_out);
    // 5) selective scan: ys = scan(h) + D0*xs  (state in VGPRs)
    scan_kernel<<<dim3(BATCH, DINNER / 64), 256, 0, stream>>>(conv_out, dtbuf, A_log, Dp, ysbuf);
    // 6) ygate = rmsnorm(ys * silu(z)) * gated_norm_w -> bf16
    gate_norm_kernel<<<MTOT, 256, 0, stream>>>(ysbuf, zxbcdt, gated_w, ygate_bf);
    // 7) resid = ygate @ out_proj_w^T + x   (16384 x 512, K=1024)
    gemm_wmma_kernel<1><<<dim3(DMODEL / 128, MTOT / 128), 256, 0, stream>>>(
        ygate_bf, w_out_bf, MTOT, DMODEL, DINNER, resid, nullptr, x);
    // 8) h2 = rmsnorm(resid, norm2_w) -> bf16
    rmsnorm512_bf16_kernel<<<MTOT, 256, 0, stream>>>(resid, norm2_w, h2_bf);
    // 9) gelu(h2 @ mlp_w1^T) -> bf16   (16384 x 2048, K=512)
    gemm_wmma_kernel<2><<<dim3((4 * DMODEL) / 128, MTOT / 128), 256, 0, stream>>>(
        h2_bf, w_m1_bf, MTOT, 4 * DMODEL, DMODEL, nullptr, gelu_bf, nullptr);
    // 10) y = gelu @ mlp_w2^T -> d_out (f32)   (16384 x 512, K=2048)
    gemm_wmma_kernel<0><<<dim3(DMODEL / 128, MTOT / 128), 256, 0, stream>>>(
        gelu_bf, w_m2_bf, MTOT, DMODEL, 4 * DMODEL, out, nullptr, nullptr);
    // 11) rnn_state passthrough
    copy_f32_kernel<<<(BATCH * DMODEL + 255) / 256, 256, 0, stream>>>(
        rnn_state, out + (size_t)MTOT * DMODEL, BATCH * DMODEL);
}
